// NNConvModel_50328426774919
// MI455X (gfx1250) — matrix operand order repaired and checked by reading, not compiled, not run on hardware
//
#include <hip/hip_runtime.h>
#include <hip/hip_bf16.h>

typedef __attribute__((ext_vector_type(16))) _Float16 v16h;
typedef __attribute__((ext_vector_type(8)))  float    v8f;
typedef unsigned int u32;

#define N_NODES 20000
#define N_EDGES 100000
#define LEAK 0.1f

// ---------------------------------------------------------------------------
// helpers
// ---------------------------------------------------------------------------
__device__ inline v8f zero8() {
  v8f z = {0.f,0.f,0.f,0.f,0.f,0.f,0.f,0.f};
  return z;
}

__device__ inline float lrelu(float v) { return v > 0.f ? v : LEAK * v; }

// Load a pre-packed B tile (32 lanes x 8 dwords, f16 pairs) -> v16h
__device__ inline v16h load_pk(const u32* base, int lane) {
  const u32* p = base + lane * 8;
  union { u32 u[8]; v16h h; } cv;
  uint4 a = *(const uint4*)p;
  uint4 b = *(const uint4*)(p + 4);
  cv.u[0]=a.x; cv.u[1]=a.y; cv.u[2]=a.z; cv.u[3]=a.w;
  cv.u[4]=b.x; cv.u[5]=b.y; cv.u[6]=b.z; cv.u[7]=b.w;
  return cv.h;
}

// Load A operand (16x32 f16) from row-major f32 LDS buffer, stride ld.
// A layout: lanes 0-15 (row M=lane): v0-3 K=0..7, v4-7 K=16..23
//           lanes 16-31 (row M=lane-16): v0-3 K=8..15, v4-7 K=24..31
__device__ inline v16h load_A_lds(const float* s, int ld, int lane) {
  int half = lane >> 4, row = lane & 15;
  v16h a;
#pragma unroll
  for (int j = 0; j < 8; ++j) {
    int k0 = (j < 4 ? 0 : 16) + half * 8 + (j & 3) * 2;
    a[2*j]   = (_Float16)s[row * ld + k0];
    a[2*j+1] = (_Float16)s[row * ld + k0 + 1];
  }
  return a;
}

// ---------------------------------------------------------------------------
// utility kernels
// ---------------------------------------------------------------------------
__global__ void zero_ws_kernel(float* p, int n) {
  int i = blockIdx.x * blockDim.x + threadIdx.x;
  if (i < n) p[i] = 0.f;
}

// Pack row-major f32 weight [K][N] into WMMA B-operand layout, f16 pairs.
// Tile (kc, nt) covers K in [kc*32, kc*32+32) x N in [nt*16, nt*16+16).
// dword index: ((kc*NT + nt)*32 + lane)*8 + j
// lane: N = nt*16 + (lane&15); K pair = kc*32 + (j<4?0:16) + (lane>>4)*8 + (j&3)*2
__global__ void pack_b_kernel(const float* w, int K, int N, int KC, u32* out) {
  int NT = N / 16;
  int total = KC * NT * 256;
  int idx = blockIdx.x * blockDim.x + threadIdx.x;
  if (idx >= total) return;
  int j = idx & 7;
  int lane = (idx >> 3) & 31;
  int t = idx >> 8;
  int kc = t / NT, nt = t % NT;
  int k0 = kc * 32 + (j < 4 ? 0 : 16) + ((lane >> 4) * 8) + (j & 3) * 2;
  int n = nt * 16 + (lane & 15);
  float f0 = (k0     < K) ? w[(size_t)k0 * N + n]       : 0.f;
  float f1 = (k0 + 1 < K) ? w[(size_t)(k0 + 1) * N + n] : 0.f;
  union { _Float16 h[2]; u32 u; } cv;
  cv.h[0] = (_Float16)f0;
  cv.h[1] = (_Float16)f1;
  out[idx] = cv.u;
}

// BatchNorm pass 1: per-channel sum & sum-of-squares (c <= 16)
__global__ void bn_stats_kernel(const float* v, int n, int c,
                                float* sums, float* sqs) {
  int ch = threadIdx.x & 15;
  int r0 = blockIdx.x * (blockDim.x >> 4) + (threadIdx.x >> 4);
  int rs = gridDim.x * (blockDim.x >> 4);
  if (ch >= c) return;
  float s = 0.f, q = 0.f;
  for (int r = r0; r < n; r += rs) {
    float val = v[(size_t)r * c + ch];
    s += val; q += val * val;
  }
  atomicAdd(&sums[ch], s);
  atomicAdd(&sqs[ch], q);
}

// BatchNorm pass 2: normalize with batch stats (biased var), affine
__global__ void bn_apply_kernel(const float* v, int n, int c,
                                const float* sums, const float* sqs,
                                const float* g, const float* b, float* out) {
  int idx = blockIdx.x * blockDim.x + threadIdx.x;
  if (idx >= n * c) return;
  int ch = idx % c;
  float inv_n = 1.0f / (float)n;
  float m = sums[ch] * inv_n;
  float var = sqs[ch] * inv_n - m * m;
  out[idx] = (v[idx] - m) * rsqrtf(var + 1e-5f) * g[ch] + b[ch];
}

// x_out = agg + x_in @ root + bias
__global__ void node_update_kernel(const float* agg, const float* xin,
                                   const float* root, const float* bias,
                                   float* xout, int n, int cin, int cout) {
  int idx = blockIdx.x * blockDim.x + threadIdx.x;
  if (idx >= n * cout) return;
  int node = idx / cout, o = idx % cout;
  float acc = agg[idx] + bias[o];
  const float* xr = xin + (size_t)node * cin;
#pragma unroll 8
  for (int k = 0; k < cin; ++k) acc += xr[k] * root[(size_t)k * cout + o];
  xout[idx] = acc;
}

// ---------------------------------------------------------------------------
// Fused NNConv edge kernel: per 16-edge tile,
//   h = lrelu(e @ w1 + b1)                 (HN outputs, WMMA)
//   for i in 0..CIN-1:  W_i = lrelu(h @ w2[:, i*COUT : ...] + b2)   (WMMA)
//                       msg += x_src[:, i] * W_i                    (VALU)
//   atomicAdd msg into agg[dst]
// One wave (32 threads) per block; EXEC all-1s around WMMA.
// ---------------------------------------------------------------------------
template <int CIN, int COUT, int HN>
__global__ void edge_conv_kernel(const float* __restrict__ en,
                                 const float* __restrict__ xin,
                                 const int* __restrict__ src,
                                 const int* __restrict__ dst,
                                 const u32* __restrict__ w1pk, const float* __restrict__ b1,
                                 const u32* __restrict__ w2pk, const float* __restrict__ b2,
                                 float* __restrict__ agg, int n_edges) {
  __shared__ float s_h[16 * 32];
  __shared__ float s_x[16 * 32];
  __shared__ int s_src[16], s_dst[16];

  int lane = threadIdx.x;
  int half = lane >> 4, rrow = lane & 15;
  int nTiles = n_edges / 16;

  // zero h scratch once (covers K-padding for HN==16 case)
  for (int t = lane; t < 16 * 32; t += 32) s_h[t] = 0.f;
  __syncthreads();

  for (int tile = blockIdx.x; tile < nTiles; tile += gridDim.x) {
    int eb = tile * 16;
    if (lane < 16) { s_src[lane] = src[eb + lane]; s_dst[lane] = dst[eb + lane]; }

    // ---- A operand: edge-feature tile (K=10, zero-padded to 32) ----
    const float* erow = en + (size_t)(eb + rrow) * 10;
    v16h a_e;
#pragma unroll
    for (int j = 0; j < 8; ++j) {
      int k0 = (j < 4 ? 0 : 16) + half * 8 + (j & 3) * 2;
      float f0 = (k0     < 10) ? erow[k0]     : 0.f;
      float f1 = (k0 + 1 < 10) ? erow[k0 + 1] : 0.f;
      a_e[2*j]   = (_Float16)f0;
      a_e[2*j+1] = (_Float16)f1;
    }

    // ---- h = lrelu(e @ w1 + b1) -> LDS [16][32] row-major ----
#pragma unroll
    for (int nt = 0; nt < HN / 16; ++nt) {
      v16h b = load_pk(w1pk + nt * 256, lane);
      v8f c = zero8();
      c = __builtin_amdgcn_wmma_f32_16x16x32_f16(false, a_e, false, b,
                                                 (short)0, c, false, false);
      float bias = b1[nt * 16 + rrow];
#pragma unroll
      for (int r = 0; r < 8; ++r) {
        float v = lrelu(c[r] + bias);
        s_h[(r + 8 * half) * 32 + nt * 16 + rrow] = v;   // (M, N)
      }
    }
    __syncthreads();

    v16h a_h = load_A_lds(s_h, 32, lane);   // resident A for whole i-loop

    // ---- gather x[src] tile into LDS [16][CIN] ----
    for (int t = lane; t < 16 * CIN; t += 32) {
      int m = t / CIN, i = t % CIN;
      s_x[m * CIN + i] = xin[(size_t)s_src[m] * CIN + i];
    }
    __syncthreads();

    v8f msg[COUT / 16];
#pragma unroll
    for (int nt = 0; nt < COUT / 16; ++nt) msg[nt] = zero8();

    for (int i = 0; i < CIN; ++i) {
      float xs[8];
#pragma unroll
      for (int r = 0; r < 8; ++r) xs[r] = s_x[(r + 8 * half) * CIN + i];
#pragma unroll
      for (int nt = 0; nt < COUT / 16; ++nt) {
        int tIdx = i * (COUT / 16) + nt;
        v16h b = load_pk(w2pk + tIdx * 256, lane);
        v8f c = zero8();
        c = __builtin_amdgcn_wmma_f32_16x16x32_f16(false, a_h, false, b,
                                                   (short)0, c, false, false);
        float bias = b2[tIdx * 16 + rrow];
#pragma unroll
        for (int r = 0; r < 8; ++r) {
          float w = lrelu(c[r] + bias);
          msg[nt][r] += xs[r] * w;
        }
      }
    }

    // ---- scatter-add to agg[dst] ----
#pragma unroll
    for (int nt = 0; nt < COUT / 16; ++nt) {
#pragma unroll
      for (int r = 0; r < 8; ++r) {
        int M = r + 8 * half;
        atomicAdd(&agg[(size_t)s_dst[M] * COUT + nt * 16 + rrow], msg[nt][r]);
      }
    }
    __syncthreads();
  }
}

// ---------------------------------------------------------------------------
// Fused edge MLP: cat(x2[src], x2[dst], e) [16 x 138] -> 64 -> 32 -> 16 -> 8 -> 2
// First three layers via WMMA, last two tiny layers in VALU.
// ---------------------------------------------------------------------------
__global__ void edge_mlp_kernel(const float* __restrict__ x2,
                                const float* __restrict__ en,
                                const int* __restrict__ src,
                                const int* __restrict__ dst,
                                const u32* w1pk, const float* b1,
                                const u32* w2pk, const float* b2,
                                const u32* w3pk, const float* b3,
                                const float* w4, const float* b4,
                                const float* w5, const float* b5,
                                float* __restrict__ out, int n_edges) {
  __shared__ float s_in[16 * 160];   // 138 padded to 160 (5 K-chunks of 32)
  __shared__ float s_h[16 * 64];
  __shared__ int s_src[16], s_dst[16];

  int lane = threadIdx.x;
  int half = lane >> 4, rrow = lane & 15;
  int nTiles = n_edges / 16;

  for (int t = lane; t < 16 * 160; t += 32) s_in[t] = 0.f;  // zero K-padding
  __syncthreads();

  for (int tile = blockIdx.x; tile < nTiles; tile += gridDim.x) {
    int eb = tile * 16;
    if (lane < 16) { s_src[lane] = src[eb + lane]; s_dst[lane] = dst[eb + lane]; }
    __syncthreads();

    // build concat input
    for (int t = lane; t < 16 * 64; t += 32) {
      int m = t >> 6, k = t & 63;
      s_in[m * 160 + k]      = x2[(size_t)s_src[m] * 64 + k];
      s_in[m * 160 + 64 + k] = x2[(size_t)s_dst[m] * 64 + k];
    }
    for (int t = lane; t < 16 * 10; t += 32) {
      int m = t / 10, k = t % 10;
      s_in[m * 160 + 128 + k] = en[(size_t)(eb + m) * 10 + k];
    }
    __syncthreads();

    // layer 1: 138 -> 64 (KC=5, NT=4)
    v8f acc[4];
#pragma unroll
    for (int nt = 0; nt < 4; ++nt) acc[nt] = zero8();
    for (int kc = 0; kc < 5; ++kc) {
      v16h a = load_A_lds(s_in + kc * 32, 160, lane);
#pragma unroll
      for (int nt = 0; nt < 4; ++nt) {
        v16h b = load_pk(w1pk + (kc * 4 + nt) * 256, lane);
        acc[nt] = __builtin_amdgcn_wmma_f32_16x16x32_f16(false, a, false, b,
                                                         (short)0, acc[nt], false, false);
      }
    }
#pragma unroll
    for (int nt = 0; nt < 4; ++nt) {
      float bias = b1[nt * 16 + rrow];
#pragma unroll
      for (int r = 0; r < 8; ++r)
        s_h[(r + 8 * half) * 64 + nt * 16 + rrow] = lrelu(acc[nt][r] + bias);
    }
    __syncthreads();

    // layer 2: 64 -> 32 (KC=2, NT=2)
    v8f a2[2];
#pragma unroll
    for (int nt = 0; nt < 2; ++nt) a2[nt] = zero8();
    for (int kc = 0; kc < 2; ++kc) {
      v16h a = load_A_lds(s_h + kc * 32, 64, lane);
#pragma unroll
      for (int nt = 0; nt < 2; ++nt) {
        v16h b = load_pk(w2pk + (kc * 2 + nt) * 256, lane);
        a2[nt] = __builtin_amdgcn_wmma_f32_16x16x32_f16(false, a, false, b,
                                                        (short)0, a2[nt], false, false);
      }
    }
    __syncthreads();
#pragma unroll
    for (int nt = 0; nt < 2; ++nt) {
      float bias = b2[nt * 16 + rrow];
#pragma unroll
      for (int r = 0; r < 8; ++r)
        s_h[(r + 8 * half) * 32 + nt * 16 + rrow] = lrelu(a2[nt][r] + bias);
    }
    __syncthreads();

    // layer 3: 32 -> 16 (1 WMMA)
    {
      v16h a = load_A_lds(s_h, 32, lane);
      v16h b = load_pk(w3pk, lane);
      v8f c = zero8();
      c = __builtin_amdgcn_wmma_f32_16x16x32_f16(false, a, false, b,
                                                 (short)0, c, false, false);
      __syncthreads();
      float bias = b3[rrow];
#pragma unroll
      for (int r = 0; r < 8; ++r)
        s_h[(r + 8 * half) * 16 + rrow] = lrelu(c[r] + bias);
    }
    __syncthreads();

    // layers 4-5: 16 -> 8 -> 2, one edge row per lane (lanes 0..15)
    if (lane < 16) {
      float h4[8];
#pragma unroll
      for (int j = 0; j < 8; ++j) {
        float a = b4[j];
#pragma unroll
        for (int k = 0; k < 16; ++k) a += s_h[lane * 16 + k] * w4[k * 8 + j];
        h4[j] = lrelu(a);
      }
#pragma unroll
      for (int o = 0; o < 2; ++o) {
        float a = b5[o];
#pragma unroll
        for (int j = 0; j < 8; ++j) a += h4[j] * w5[j * 2 + o];
        out[(size_t)(eb + lane) * 2 + o] = a;
      }
    }
    __syncthreads();
  }
}

// ---------------------------------------------------------------------------
// launch
// ---------------------------------------------------------------------------
extern "C" void kernel_launch(void* const* d_in, const int* in_sizes, int n_in,
                              void* d_out, int out_size, void* d_ws, size_t ws_size,
                              hipStream_t stream) {
  const float* x        = (const float*)d_in[0];
  const float* e        = (const float*)d_in[1];
  const int*   ei       = (const int*)d_in[2];
  const float* bn_ng    = (const float*)d_in[4];
  const float* bn_nb    = (const float*)d_in[5];
  const float* bn_eg    = (const float*)d_in[6];
  const float* bn_eb    = (const float*)d_in[7];
  const float* nn1_w1   = (const float*)d_in[8];
  const float* nn1_b1   = (const float*)d_in[9];
  const float* nn1_w2   = (const float*)d_in[10];
  const float* nn1_b2   = (const float*)d_in[11];
  const float* nn2_w1   = (const float*)d_in[12];
  const float* nn2_b1   = (const float*)d_in[13];
  const float* nn2_w2   = (const float*)d_in[14];
  const float* nn2_b2   = (const float*)d_in[15];
  const float* l1_root  = (const float*)d_in[16];
  const float* l1_bias  = (const float*)d_in[17];
  const float* l2_root  = (const float*)d_in[18];
  const float* l2_bias  = (const float*)d_in[19];
  const float* mlp_w1   = (const float*)d_in[20];
  const float* mlp_b1   = (const float*)d_in[21];
  const float* mlp_w2   = (const float*)d_in[22];
  const float* mlp_b2   = (const float*)d_in[23];
  const float* mlp_w3   = (const float*)d_in[24];
  const float* mlp_b3   = (const float*)d_in[25];
  const float* mlp_w4   = (const float*)d_in[26];
  const float* mlp_b4   = (const float*)d_in[27];
  const float* mlp_w5   = (const float*)d_in[28];
  const float* mlp_b5   = (const float*)d_in[29];

  const int* src = ei;
  const int* dst = ei + N_EDGES;

  // workspace layout (float units)
  float* ws = (float*)d_ws;
  const size_t f_stats = 0;                           // 64 (sx16 qx16 se16 qe16)
  const size_t f_agg1  = 64;                          // 20000*32
  const size_t f_agg2  = f_agg1 + 640000;             // 20000*64
  const size_t f_xn    = f_agg2 + 1280000;            // 20000*16
  const size_t f_en    = f_xn + 320000;               // 100000*10
  const size_t f_x1    = f_en + 1000000;              // 20000*32
  const size_t f_x2    = f_x1 + 640000;               // 20000*64
  const size_t f_pk    = f_x2 + 1280000;              // packed weights (u32)
  const size_t pk_total = 48128;
  const size_t need_bytes = (f_pk + pk_total) * 4;
  if (ws_size < need_bytes) return;

  float* sx = ws + f_stats;      float* qx = sx + 16;
  float* se = qx + 16;           float* qe = se + 16;
  float* agg1 = ws + f_agg1;
  float* agg2 = ws + f_agg2;
  float* xn   = ws + f_xn;
  float* en   = ws + f_en;
  float* x1   = ws + f_x1;
  float* x2   = ws + f_x2;
  u32* pk = (u32*)(ws + f_pk);
  u32* pk_nn1w1 = pk;                     // 1 tile   = 256
  u32* pk_nn1w2 = pk_nn1w1 + 256;         // 32 tiles = 8192
  u32* pk_nn2w1 = pk_nn1w2 + 8192;        // 2 tiles  = 512
  u32* pk_nn2w2 = pk_nn2w1 + 512;         // 128 tiles= 32768
  u32* pk_mlpw1 = pk_nn2w2 + 32768;       // 20 tiles = 5120
  u32* pk_mlpw2 = pk_mlpw1 + 5120;        // 4 tiles  = 1024
  u32* pk_mlpw3 = pk_mlpw2 + 1024;        // 1 tile   = 256

  // 1) zero stats + aggregation buffers (fresh every call: deterministic)
  {
    int n = (int)(64 + 640000 + 1280000);
    zero_ws_kernel<<<(n + 255) / 256, 256, 0, stream>>>(ws, n);
  }

  // 2) pack weights into WMMA B layout (f16)
  pack_b_kernel<<<1, 256, 0, stream>>>(nn1_w1, 10, 16, 1, pk_nn1w1);
  pack_b_kernel<<<32, 256, 0, stream>>>(nn1_w2, 16, 512, 1, pk_nn1w2);
  pack_b_kernel<<<2, 256, 0, stream>>>(nn2_w1, 10, 32, 1, pk_nn2w1);
  pack_b_kernel<<<128, 256, 0, stream>>>(nn2_w2, 32, 2048, 1, pk_nn2w2);
  pack_b_kernel<<<20, 256, 0, stream>>>(mlp_w1, 138, 64, 5, pk_mlpw1);
  pack_b_kernel<<<4, 256, 0, stream>>>(mlp_w2, 64, 32, 2, pk_mlpw2);
  pack_b_kernel<<<1, 256, 0, stream>>>(mlp_w3, 32, 16, 1, pk_mlpw3);

  // 3) BatchNorm (training-mode batch stats)
  bn_stats_kernel<<<256, 256, 0, stream>>>(x, N_NODES, 16, sx, qx);
  bn_stats_kernel<<<256, 256, 0, stream>>>(e, N_EDGES, 10, se, qe);
  bn_apply_kernel<<<(N_NODES * 16 + 255) / 256, 256, 0, stream>>>(
      x, N_NODES, 16, sx, qx, bn_ng, bn_nb, xn);
  bn_apply_kernel<<<(N_EDGES * 10 + 255) / 256, 256, 0, stream>>>(
      e, N_EDGES, 10, se, qe, bn_eg, bn_eb, en);

  const int nTiles = N_EDGES / 16;  // 6250

  // 4) NNConv layer 1 (16 -> 32)
  edge_conv_kernel<16, 32, 16><<<nTiles, 32, 0, stream>>>(
      en, xn, src, dst, pk_nn1w1, nn1_b1, pk_nn1w2, nn1_b2, agg1, N_EDGES);
  node_update_kernel<<<(N_NODES * 32 + 255) / 256, 256, 0, stream>>>(
      agg1, xn, l1_root, l1_bias, x1, N_NODES, 16, 32);

  // 5) NNConv layer 2 (32 -> 64)
  edge_conv_kernel<32, 64, 32><<<nTiles, 32, 0, stream>>>(
      en, x1, src, dst, pk_nn2w1, nn2_b1, pk_nn2w2, nn2_b2, agg2, N_EDGES);
  node_update_kernel<<<(N_NODES * 64 + 255) / 256, 256, 0, stream>>>(
      agg2, x1, l2_root, l2_bias, x2, N_NODES, 32, 64);

  // 6) edge-prediction MLP
  edge_mlp_kernel<<<nTiles, 32, 0, stream>>>(
      x2, en, src, dst, pk_mlpw1, mlp_b1, pk_mlpw2, mlp_b2, pk_mlpw3, mlp_b3,
      mlp_w4, mlp_b4, mlp_w5, mlp_b5, (float*)d_out, N_EDGES);
}
